// PredictionHead_42803644072571
// MI455X (gfx1250) — compile-verified, matrix operand order checked by reading, not run on hardware
//
#include <hip/hip_runtime.h>

// ---------------------------------------------------------------------------
// PredictionHead on MI455X (gfx1250, wave32):
//   logits = relu(H @ W1 + b1) @ W2 + b2      (f16 WMMA, f32 accumulate)
//   probs  = segment_softmax(logits, batch)   (segments contiguous: batch sorted)
// Weights are staged into per-wave LDS double-buffers by the Tensor Data Mover
// (tensor_load_to_lds, TENSORcnt), one N-tile ahead of the WMMA pipeline.
// ---------------------------------------------------------------------------

typedef __attribute__((ext_vector_type(16))) _Float16 v16h;
typedef __attribute__((ext_vector_type(8)))  _Float16 v8h;
typedef __attribute__((ext_vector_type(8)))  float    v8f;
typedef __attribute__((ext_vector_type(4)))  unsigned int v4u;
typedef __attribute__((ext_vector_type(8)))  int      v8i;
typedef __attribute__((ext_vector_type(4)))  int      v4i;

#define N_NODES      500000
#define FEAT_DIM     256
#define NUM_CLASSES  128
#define NUM_SEG      2048
#define M_TILES      (N_NODES / 16)          // 31250 (exact)
#define WAVES_PER_BLOCK 2
#define GEMM_BLOCKS  (M_TILES / WAVES_PER_BLOCK) // 15625 (exact)

#define W1_FRAG_ELEMS (16 * 8 * 512)         // 65536 f16  (16 N-tiles x 8 K-steps)
#define W2_FRAG_ELEMS (8 * 8 * 512)          // 32768 f16  (8 N-tiles x 8 K-steps)
#define NTILE_ELEMS   4096                   // 8 fragments * 512 f16 = 8 KB

#if defined(__HIP_DEVICE_COMPILE__) && __has_builtin(__builtin_amdgcn_tensor_load_to_lds)
#define HAVE_TDM 1
#else
#define HAVE_TDM 0
#endif

union F16x16 {
  v16h v;
  v8h  h[2];
  _Float16 e[16];
};

__device__ __forceinline__ void cvt8(F16x16& d, int base, float4 x, float4 y) {
  d.e[base + 0] = (_Float16)x.x; d.e[base + 1] = (_Float16)x.y;
  d.e[base + 2] = (_Float16)x.z; d.e[base + 3] = (_Float16)x.w;
  d.e[base + 4] = (_Float16)y.x; d.e[base + 5] = (_Float16)y.y;
  d.e[base + 6] = (_Float16)y.z; d.e[base + 7] = (_Float16)y.w;
}

#if HAVE_TDM
// 1-D TDM copy of one 8 KB weight N-tile (1024 x 8-byte elements) into LDS.
// D# per CDNA5 ISA: group0 {count=1, lds_addr, global_addr,
// type=2}; group1 {data_size=3(8B), tensor_dim0=1024, tensor_dim1=1,
// tile_dim0=1024, tensor_dim0_stride=1024}; groups 2/3 zero (<=2D tensor).
// This toolchain exposes the 6-arg builtin: (v4u, v8i, v4i, v4i, v8i, cpol).
__device__ __forceinline__ void tdm_stage_8kb(const _Float16* gsrc, void* lds_dst) {
  const unsigned long long ga = (unsigned long long)gsrc;
  const unsigned lds_off = (unsigned)(unsigned long long)lds_dst; // addr[31:0] = LDS byte offset
  v4u g0;
  g0.x = 1u;                                   // count=1 (valid descriptor)
  g0.y = lds_off;                              // lds_addr (bytes)
  g0.z = (unsigned)ga;                         // global_addr[31:0]
  g0.w = (unsigned)(ga >> 32) | (2u << 30);    // global_addr[56:32] | type=2
  v8i g1;
  g1[0] = 0x00030000;                          // data_size = 3 -> 8 bytes
  g1[1] = (int)(1024u << 16);                  // tensor_dim0[15:0] = 1024
  g1[2] = (int)(1u << 16);                     // tensor_dim0 hi=0, tensor_dim1 = 1
  g1[3] = (int)(1024u << 16);                  // tile_dim0 = 1024
  g1[4] = 0;                                   // tile_dim1/tile_dim2 unused
  g1[5] = 1024;                                // tensor_dim0_stride = 1024
  g1[6] = 0;
  g1[7] = 0;
  const v4i z4 = {0, 0, 0, 0};
  const v8i z8 = {0, 0, 0, 0, 0, 0, 0, 0};
  __builtin_amdgcn_tensor_load_to_lds(g0, g1, z4, z4, z8, 0);
}
#endif

// ---------------------------------------------------------------------------
// Pack W1/W2 (f32 row-major, [K][N]) into per-lane WMMA B-fragment layout:
// fragment f = n_tile*8 + k_step holds the 32x16 f16 slice; lane l owns 16
// contiguous halfwords at frag*512 + l*16.  Per-lane element j maps to
//   col = n_tile*16 + (l&15)
//   k   = k_step*32 + (l>>4)*8 + j + (j>=8 ? 8 : 0)
// mirroring the 16-bit A-matrix striping from CDNA5 ISA.
// ---------------------------------------------------------------------------
__global__ void pack_weights_kernel(const float* __restrict__ W1,
                                    const float* __restrict__ W2,
                                    _Float16* __restrict__ wsh) {
  int idx = blockIdx.x * blockDim.x + threadIdx.x;
  if (idx >= W1_FRAG_ELEMS + W2_FRAG_ELEMS) return;
  const bool isW1 = idx < W1_FRAG_ELEMS;
  const int e  = isW1 ? idx : (idx - W1_FRAG_ELEMS);
  const int f  = e >> 9;         // fragment index
  const int l  = (e >> 4) & 31;  // lane
  const int j  = e & 15;         // element within lane
  const int n  = f >> 3;         // N-tile
  const int kk = f & 7;          // K-step
  const int col = n * 16 + (l & 15);
  const int k   = kk * 32 + (l >> 4) * 8 + j + ((j >= 8) ? 8 : 0);
  const float v = isW1 ? W1[k * FEAT_DIM + col] : W2[k * NUM_CLASSES + col];
  wsh[idx] = (_Float16)v;
}

// ---------------------------------------------------------------------------
// Fused MLP: one 16-row tile per wave. H read once from HBM; intermediate h
// ([16 x 256] f16) lives only in LDS. Weight fragments arrive via TDM double
// buffering; all matrix math via v_wmma_f32_16x16x32_f16.
// ---------------------------------------------------------------------------
__global__ void __launch_bounds__(64)
mlp_wmma_kernel(const float* __restrict__ H,
                const float* __restrict__ b1,
                const float* __restrict__ b2,
                const _Float16* __restrict__ wsh,
                float* __restrict__ logits) {
  __shared__ __align__(16) _Float16 hbuf[WAVES_PER_BLOCK][16][FEAT_DIM];   // 16 KB
#if HAVE_TDM
  __shared__ __align__(16) _Float16 wstage[WAVES_PER_BLOCK][2][NTILE_ELEMS]; // 32 KB
#endif

  const int wv    = threadIdx.x >> 5;
  const int l     = threadIdx.x & 31;
  const int tile  = blockIdx.x * WAVES_PER_BLOCK + wv;   // always < M_TILES
  const int m0    = tile * 16;
  const int khalf = l >> 4;       // 0: lanes 0-15, 1: lanes 16-31
  const int nl    = l & 15;

  // ---- A fragments: 16 rows of H, K=0..255, f32 -> f16 in WMMA A layout ----
  F16x16 a[8];
#if HAVE_TDM
  // Kick off staging of W1 N-tile 0 while the H loads are in flight.
  tdm_stage_8kb(wsh, &wstage[wv][0][0]);
#endif
  {
    const float* rowp = H + (size_t)(m0 + nl) * FEAT_DIM;
    #pragma unroll
    for (int kk = 0; kk < 8; ++kk) {
      const int c0 = kk * 32 + khalf * 8;
      const float4 f0 = *(const float4*)(rowp + c0);
      const float4 f1 = *(const float4*)(rowp + c0 + 4);
      const float4 f2 = *(const float4*)(rowp + c0 + 16);
      const float4 f3 = *(const float4*)(rowp + c0 + 20);
      cvt8(a[kk], 0, f0, f1);
      cvt8(a[kk], 8, f2, f3);
    }
  }

  // ---- GEMM1: h = relu(H @ W1 + b1) -> LDS (f16, row-major) ----
  #pragma unroll
  for (int n = 0; n < 16; ++n) {
#if HAVE_TDM
    if (n + 1 < 16) {
      tdm_stage_8kb(wsh + (n + 1) * NTILE_ELEMS, &wstage[wv][(n + 1) & 1][0]);
      __builtin_amdgcn_s_wait_tensorcnt(1);   // N-tile n resident
    } else {
      __builtin_amdgcn_s_wait_tensorcnt(0);
    }
    const _Float16* wb = &wstage[wv][n & 1][0];
#else
    const _Float16* wb = wsh + n * NTILE_ELEMS;
#endif
    v8f acc = {};
    #pragma unroll
    for (int kk = 0; kk < 8; ++kk) {
      F16x16 b;
      const v8h* bp = (const v8h*)(wb + (kk << 9) + (l << 4));
      b.h[0] = bp[0];
      b.h[1] = bp[1];
      acc = __builtin_amdgcn_wmma_f32_16x16x32_f16(
          false, a[kk].v, false, b.v, (short)0, acc, false, false);
    }
    const float bias = b1[n * 16 + nl];
    #pragma unroll
    for (int r = 0; r < 8; ++r) {
      float x = acc[r] + bias;             // C layout: VGPR r -> M = r + khalf*8
      x = x > 0.f ? x : 0.f;               // ReLU
      hbuf[wv][r + khalf * 8][n * 16 + nl] = (_Float16)x;
    }
  }

  __syncthreads();   // both waves full -> safe; orders LDS writes before reads

  // ---- reload h as A fragments for GEMM2 (ds_load_b128) ----
#if HAVE_TDM
  tdm_stage_8kb(wsh + W1_FRAG_ELEMS, &wstage[wv][0][0]);  // W2 N-tile 0
#endif
  #pragma unroll
  for (int kk = 0; kk < 8; ++kk) {
    const int c0 = kk * 32 + khalf * 8;
    a[kk].h[0] = *(const v8h*)&hbuf[wv][nl][c0];
    a[kk].h[1] = *(const v8h*)&hbuf[wv][nl][c0 + 16];
  }

  // ---- GEMM2: logits = h @ W2 + b2 ----
  const _Float16* w2f = wsh + W1_FRAG_ELEMS;
  #pragma unroll
  for (int n = 0; n < 8; ++n) {
#if HAVE_TDM
    if (n + 1 < 8) {
      tdm_stage_8kb(w2f + (n + 1) * NTILE_ELEMS, &wstage[wv][(n + 1) & 1][0]);
      __builtin_amdgcn_s_wait_tensorcnt(1);
    } else {
      __builtin_amdgcn_s_wait_tensorcnt(0);
    }
    const _Float16* wb = &wstage[wv][n & 1][0];
#else
    const _Float16* wb = w2f + n * NTILE_ELEMS;
#endif
    v8f acc = {};
    #pragma unroll
    for (int kk = 0; kk < 8; ++kk) {
      F16x16 b;
      const v8h* bp = (const v8h*)(wb + (kk << 9) + (l << 4));
      b.h[0] = bp[0];
      b.h[1] = bp[1];
      acc = __builtin_amdgcn_wmma_f32_16x16x32_f16(
          false, a[kk].v, false, b.v, (short)0, acc, false, false);
    }
    const float bias = b2[n * 16 + nl];
    #pragma unroll
    for (int r = 0; r < 8; ++r) {
      logits[(size_t)(m0 + r + khalf * 8) * NUM_CLASSES + n * 16 + nl] =
          acc[r] + bias;
    }
  }
}

// ---------------------------------------------------------------------------
// Segment softmax: batch is sorted, so each segment is a contiguous row run.
// One block per segment, one thread per class column. Boundaries via binary
// search; three coalesced passes (max, exp+sum, normalize). No atomics.
// ---------------------------------------------------------------------------
__global__ void __launch_bounds__(128)
segment_softmax_kernel(const int* __restrict__ batch,
                       const float* __restrict__ logits,
                       float* __restrict__ probs) {
  const int s = blockIdx.x;
  const int c = threadIdx.x;

  int lo = 0, hi = N_NODES;
  while (lo < hi) { int mid = (lo + hi) >> 1; if (batch[mid] < s) lo = mid + 1; else hi = mid; }
  const int start = lo;
  hi = N_NODES;
  while (lo < hi) { int mid = (lo + hi) >> 1; if (batch[mid] < s + 1) lo = mid + 1; else hi = mid; }
  const int end = lo;

  float mx = -3.402823466e38f;
  for (int i = start; i < end; ++i)
    mx = fmaxf(mx, logits[(size_t)i * NUM_CLASSES + c]);

  float sum = 0.f;
  for (int i = start; i < end; ++i) {
    const float e = __expf(logits[(size_t)i * NUM_CLASSES + c] - mx);
    probs[(size_t)i * NUM_CLASSES + c] = e;
    sum += e;
  }

  const float inv = 1.f / sum;
  for (int i = start; i < end; ++i)
    probs[(size_t)i * NUM_CLASSES + c] *= inv;
}

// ---------------------------------------------------------------------------
extern "C" void kernel_launch(void* const* d_in, const int* in_sizes, int n_in,
                              void* d_out, int out_size, void* d_ws, size_t ws_size,
                              hipStream_t stream) {
  // setup_inputs order: H, batch, num_segments, W1, b1, W2, b2
  const float* H     = (const float*)d_in[0];
  const int*   batch = (const int*)d_in[1];
  const float* W1    = (const float*)d_in[3];
  const float* b1    = (const float*)d_in[4];
  const float* W2    = (const float*)d_in[5];
  const float* b2    = (const float*)d_in[6];

  float* logits = (float*)d_out;
  float* probs  = logits + (size_t)N_NODES * NUM_CLASSES;
  _Float16* wsh = (_Float16*)d_ws;   // 192 KB of f16 weight fragments

  const int pack_total = W1_FRAG_ELEMS + W2_FRAG_ELEMS;   // 98304
  pack_weights_kernel<<<(pack_total + 255) / 256, 256, 0, stream>>>(W1, W2, wsh);
  mlp_wmma_kernel<<<GEMM_BLOCKS, 64, 0, stream>>>(H, b1, b2, wsh, logits);
  segment_softmax_kernel<<<NUM_SEG, 128, 0, stream>>>(batch, logits, probs);
}
